// NeuralCDE_89996744721005
// MI455X (gfx1250) — matrix operand order repaired and checked by reading, not compiled, not run on hardware
//
#include <hip/hip_runtime.h>
#include <hip/hip_bf16.h>
#include <math.h>

// Problem dims (match reference)
#define Bsz 128
#define Tsz 128
#define Dsz 32
#define Hsz 128
#define Wsz 128
#define Osz 16
#define DpC 33   // D+1
#define Ssub 2
#define NWG 8    // workgroups in the persistent grid (one 16-wide h-tile each)

typedef _Float16 h16;
typedef __attribute__((ext_vector_type(16))) _Float16 v16h;
typedef __attribute__((ext_vector_type(8)))  _Float16 v8h;
typedef __attribute__((ext_vector_type(8)))  float    v8f;

typedef __attribute__((ext_vector_type(4))) unsigned int u32x4;
typedef __attribute__((ext_vector_type(8))) int i32x8;
typedef __attribute__((ext_vector_type(4))) int i32x4;

union V16U { v16h v; v8h h[2]; };

#if defined(__has_builtin)
#if __has_builtin(__builtin_amdgcn_tensor_load_to_lds) && __has_builtin(__builtin_amdgcn_s_wait_tensorcnt)
#define USE_TDM 1
#endif
#endif
#ifndef USE_TDM
#define USE_TDM 0
#endif

// Branch-free tanh (rational approx, |err| < ~1e-6); vectorizes to v_pk_fma_f32.
__device__ inline float fast_tanh(float x) {
  const float xc = fminf(fmaxf(x, -9.0f), 9.0f);
  const float x2 = xc * xc;
  float p = -2.76076847742355e-16f;
  p = fmaf(p, x2, 2.00018790482477e-13f);
  p = fmaf(p, x2, -8.60467152213735e-11f);
  p = fmaf(p, x2, 5.12229709037114e-08f);
  p = fmaf(p, x2, 1.48572235717979e-05f);
  p = fmaf(p, x2, 6.37261928875436e-04f);
  p = fmaf(p, x2, 4.89352455891786e-03f);
  p = p * xc;
  float q = 1.19825839466702e-06f;
  q = fmaf(q, x2, 1.18534705686654e-04f);
  q = fmaf(q, x2, 2.26843463243900e-03f);
  q = fmaf(q, x2, 4.89352518554385e-03f);
  return p * __builtin_amdgcn_rcpf(q);
}

// A operand (16x32 f16): lanes 0-15 -> K {0..7,16..23}, lanes 16-31 -> K {8..15,24..31}
__device__ inline v16h load_a_f16(const h16* base, int ld, int ks) {
  const int lane = threadIdx.x & 31;
  const int m = lane & 15;
  const int kb = ((lane & 16) ? 8 : 0) + ks * 32;
  V16U r;
  r.h[0] = *(const v8h*)(base + m * ld + kb);
  r.h[1] = *(const v8h*)(base + m * ld + kb + 16);
  return r.v;
}
// B operand (32x16 f16), stored row-per-output-column (N rows, K contiguous)
__device__ inline v16h load_b_f16(const h16* base, int ld, int ks) {
  const int lane = threadIdx.x & 31;
  const int n = lane & 15;
  const int kb = ((lane & 16) ? 8 : 0) + ks * 32;
  V16U r;
  r.h[0] = *(const v8h*)(base + n * ld + kb);
  r.h[1] = *(const v8h*)(base + n * ld + kb + 16);
  return r.v;
}

#if USE_TDM
// TDM 2D tile load: 16 rows x 128 f16 (256B/row) global -> LDS.
// D# per CDNA5 ISA.
__device__ inline void tdm_load_b_panel(const h16* gptr, unsigned lds_byte_addr) {
  unsigned long long ga = (unsigned long long)(size_t)gptr;
  u32x4 g0;
  g0[0] = 1u;                                                // count=1 (user D#)
  g0[1] = lds_byte_addr;                                     // LDS dest (bytes)
  g0[2] = (unsigned)(ga & 0xffffffffu);                      // global_addr[31:0]
  g0[3] = (unsigned)((ga >> 32) & 0x01ffffffu) | (2u << 30); // addr[56:32] | type=2
  i32x8 g1;
  g1[0] = (int)(1u << 16);            // workgroup_mask=0, data_size=1 (2 bytes)
  g1[1] = (int)(128u << 16);          // tensor_dim0[15:0]=128
  g1[2] = (int)(16u << 16);           // tensor_dim1[15:0]=16
  g1[3] = (int)(128u << 16);          // tile_dim0=128
  g1[4] = (int)(16u);                 // tile_dim1=16
  g1[5] = (int)(128u);                // tensor_dim0_stride=128
  g1[6] = 0;
  g1[7] = 0;
  i32x4 gz = {0, 0, 0, 0};
#if __clang_major__ >= 23
  i32x8 gz8 = {0, 0, 0, 0, 0, 0, 0, 0};
  __builtin_amdgcn_tensor_load_to_lds(g0, g1, gz, gz, gz8, 0);
#else
  __builtin_amdgcn_tensor_load_to_lds(g0, g1, gz, gz, 0);
#endif
}
#endif

// Device-scope barrier across the NWG persistent workgroups.
__device__ inline void grid_sync(unsigned* ctr, unsigned* gen) {
  __syncthreads();
  if (threadIdx.x == 0) {
    __threadfence();                       // release this WG's stores
    unsigned g = ++(*gen);
    atomicAdd(ctr, 1u);
    while (__hip_atomic_load(ctr, __ATOMIC_RELAXED, __HIP_MEMORY_SCOPE_AGENT) <
           (unsigned)NWG * g) {
      __builtin_amdgcn_s_sleep(2);
    }
    __threadfence();                       // acquire other WGs' stores
  }
  __syncthreads();
}

// One-time prep: convert W1/W2 to f16, permute+convert W3 to [d][h] column
// order, permute b3, copy z0, reset the grid-sync counter.
__global__ void prep_kernel(const float* __restrict__ W1, const float* __restrict__ W2,
                            const float* __restrict__ W3, const float* __restrict__ b3,
                            const float* __restrict__ z0,
                            h16* W1h, h16* W2h, h16* W3p, float* b3p, float* z,
                            unsigned* sync_ctr) {
  if (blockIdx.x == 0 && threadIdx.x == 0) *sync_ctr = 0u;
  int idx = blockIdx.x * blockDim.x + threadIdx.x;
  int stride = gridDim.x * blockDim.x;
  for (int i = idx; i < Hsz * DpC * Wsz; i += stride) {
    int k = i & (Wsz - 1);
    int c = i >> 7;                 // original row h*Dp + d
    int h = c / DpC, d = c % DpC;
    W3p[((size_t)(d * Hsz + h)) * Wsz + k] = (h16)W3[i];
    if (i < Hsz * Hsz) { W1h[i] = (h16)W1[i]; W2h[i] = (h16)W2[i]; }
    if (i < Bsz * Hsz) z[i] = z0[i];
    if (i < Hsz * DpC) { int hh = i / DpC, dd = i % DpC; b3p[dd * Hsz + hh] = b3[i]; }
  }
}

// Persistent kernel: the entire CDE scan (127 intervals x 2 RK4 substeps x 4
// stages) in one launch. Each WG redundantly computes the small MLP in LDS,
// then computes its 16-wide h-slice of the dominant GEMM3 + Dp-contraction;
// k-slices are exchanged through global memory with a device-scope barrier.
__global__ void __launch_bounds__(256) cde_kernel(
    const float* __restrict__ ts, const float* __restrict__ xs,
    const h16* __restrict__ W1h, const float* __restrict__ b1,
    const h16* __restrict__ W2h, const float* __restrict__ b2,
    const h16* __restrict__ W3p, const float* __restrict__ b3p,
    const float* __restrict__ Wo, const float* __restrict__ bo,
    float* __restrict__ z, float* __restrict__ kbase,
    unsigned* sync_ctr, float* __restrict__ out) {
  __shared__ h16 zzh[Hsz * Hsz];      // zz (f16) for GEMM1; reused for a2 after GEMM2
  __shared__ h16 a1h[Hsz * Hsz];      // a1 (f16)
  __shared__ float dXs[DpC * Bsz];    // dX/dt for the current stage
  __shared__ h16 bpan[2][16 * Wsz];   // double-buffered W3p B panel (TDM dest)

  const int tid = threadIdx.x;
  const int wave = tid >> 5;
  const int lane = tid & 31;
  const int m0 = wave * 16;
  const int n = lane & 15;
  const int rb = (lane & 16) ? 8 : 0;
  const int h0 = blockIdx.x * 16;
  unsigned gen = 0;
#if USE_TDM
  const unsigned bp_addr[2] = {(unsigned)(size_t)(void*)&bpan[0][0],
                               (unsigned)(size_t)(void*)&bpan[1][0]};
#endif

  for (int t = 0; t < Tsz - 1; ++t) {
    const float t0 = ts[t], t1 = ts[t + 1];
    const float hh = t1 - t0;
    const float inv_h = 1.0f / hh;
    const float dt = hh * (1.0f / Ssub);
    for (int j = 0; j < Ssub; ++j) {
      for (int st = 0; st < 4; ++st) {
        const float coef = (st == 0) ? 0.f : ((st == 3) ? 1.f : 0.5f);
        const float s = (st == 0) ? (j * (1.f / Ssub))
                       : (st == 3) ? ((j + 1) * (1.f / Ssub))
                                   : ((j + 0.5f) * (1.f / Ssub));
        const float* kprev = kbase + ((st + 3) & 3) * (Bsz * Hsz);
        // ---- phase A: zz = z + coef*dt*k_prev (f16 to LDS) + dX/dt table ----
        const float ca = coef * dt;
        for (int i = tid; i < Bsz * Hsz; i += 256) {
          float v = z[i];
          if (st != 0) v += ca * kprev[i];
          zzh[i] = (h16)v;
        }
        {
          const float s2 = s * s;
          const float c0x = (6.f * s2 - 6.f * s) * inv_h;
          const float c1x = (6.f * s - 6.f * s2) * inv_h;
          const float cd0 = 3.f * s2 - 4.f * s + 1.f;
          const float cd1 = 3.f * s2 - 2.f * s;
          for (int i = tid; i < Bsz * DpC; i += 256) {
            int b = i / DpC, d = i % DpC;
            float x0, x1, d0, d1;
            if (d == 0) {
              x0 = t0; x1 = t1; d0 = 1.f; d1 = 1.f;
            } else {
              const float* xb = xs + (size_t)(b * Tsz) * Dsz + (d - 1);
              x0 = xb[t * Dsz];
              x1 = xb[(t + 1) * Dsz];
              d1 = (x1 - x0) * inv_h;
              d0 = (t > 0) ? (x0 - xb[(t - 1) * Dsz]) / (t0 - ts[t - 1]) : d1;
            }
            dXs[d * Bsz + b] = c0x * x0 + c1x * x1 + cd0 * d0 + cd1 * d1;
          }
        }
        __syncthreads();
        // ---- GEMM1: a1 = tanh(zz @ W1^T + b1) ----
        for (int nt = 0; nt < 8; ++nt) {
          const int n0 = nt * 16;
          v8f c = {};
          for (int ks = 0; ks < 4; ++ks) {
            v16h a = load_a_f16(zzh + m0 * Hsz, Hsz, ks);
            v16h b = load_b_f16(W1h + (size_t)n0 * Hsz, Hsz, ks);
            c = __builtin_amdgcn_wmma_f32_16x16x32_f16(false, a, false, b, (short)0, c, false, false);
          }
          const float bb = b1[n0 + n];
          for (int r = 0; r < 8; ++r)
            a1h[(m0 + r + rb) * Hsz + n0 + n] = (h16)fast_tanh(c[r] + bb);
        }
        __syncthreads();
        // ---- GEMM2: a2 = tanh(a1 @ W2^T + b2) -> reuse zzh as a2 ----
        for (int nt = 0; nt < 8; ++nt) {
          const int n0 = nt * 16;
          v8f c = {};
          for (int ks = 0; ks < 4; ++ks) {
            v16h a = load_a_f16(a1h + m0 * Hsz, Hsz, ks);
            v16h b = load_b_f16(W2h + (size_t)n0 * Hsz, Hsz, ks);
            c = __builtin_amdgcn_wmma_f32_16x16x32_f16(false, a, false, b, (short)0, c, false, false);
          }
          const float bb = b2[n0 + n];
          for (int r = 0; r < 8; ++r)
            zzh[(m0 + r + rb) * Hsz + n0 + n] = (h16)fast_tanh(c[r] + bb);
        }
        __syncthreads();
        // ---- GEMM3 + fused Dp-contraction for this WG's h-slice ----
        v16h a[4];
        for (int ks = 0; ks < 4; ++ks) a[ks] = load_a_f16(zzh + m0 * Hsz, Hsz, ks);
        float acc[8];
        for (int r = 0; r < 8; ++r) acc[r] = 0.f;
#if USE_TDM
        if (wave == 0)
          tdm_load_b_panel(W3p + (size_t)(0 * Hsz + h0) * Wsz, bp_addr[0]);
        for (int d = 0; d < DpC; ++d) {
          __builtin_amdgcn_s_wait_tensorcnt(0);   // nop for non-issuing waves
          __syncthreads();
          if (wave == 0 && d + 1 < DpC)           // prefetch next panel (overlaps WMMAs)
            tdm_load_b_panel(W3p + (size_t)((d + 1) * Hsz + h0) * Wsz,
                             bp_addr[(d + 1) & 1]);
          const h16* bp = &bpan[d & 1][0];
          v8f c = {};
          for (int ks = 0; ks < 4; ++ks) {
            v16h b = load_b_f16(bp, Wsz, ks);
            c = __builtin_amdgcn_wmma_f32_16x16x32_f16(false, a[ks], false, b, (short)0, c, false, false);
          }
          const float bb = b3p[d * Hsz + h0 + n];
          for (int r = 0; r < 8; ++r) {
            float f = fast_tanh(c[r] + bb);
            acc[r] += f * dXs[d * Bsz + m0 + r + rb];
          }
        }
#else
        for (int d = 0; d < DpC; ++d) {
          __syncthreads();
          {
            int row = tid >> 4;
            int off = (tid & 15) * 8;
            *(v8h*)(&bpan[0][0] + row * Wsz + off) =
                *(const v8h*)(W3p + (size_t)(d * Hsz + h0 + row) * Wsz + off);
          }
          __syncthreads();
          v8f c = {};
          for (int ks = 0; ks < 4; ++ks) {
            v16h b = load_b_f16(&bpan[0][0], Wsz, ks);
            c = __builtin_amdgcn_wmma_f32_16x16x32_f16(false, a[ks], false, b, (short)0, c, false, false);
          }
          const float bb = b3p[d * Hsz + h0 + n];
          for (int r = 0; r < 8; ++r) {
            float f = fast_tanh(c[r] + bb);
            acc[r] += f * dXs[d * Bsz + m0 + r + rb];
          }
        }
#endif
        float* kst = kbase + st * (Bsz * Hsz);
        for (int r = 0; r < 8; ++r)
          kst[(size_t)(m0 + r + rb) * Hsz + h0 + n] = acc[r];
        grid_sync(sync_ctr, &gen);   // publish k-slice, see all slices
      }
      // ---- RK4 combine (each WG owns a 2048-element slice of z) ----
      {
        const float cc = dt / 6.0f;
        const float* k1 = kbase;
        const float* k2 = kbase + Bsz * Hsz;
        const float* k3 = kbase + 2 * Bsz * Hsz;
        const float* k4 = kbase + 3 * Bsz * Hsz;
        for (int i = tid; i < (Bsz * Hsz) / NWG; i += 256) {
          int idx = blockIdx.x * ((Bsz * Hsz) / NWG) + i;
          z[idx] += cc * (k1[idx] + 2.f * k2[idx] + 2.f * k3[idx] + k4[idx]);
        }
      }
      grid_sync(sync_ctr, &gen);
    }
  }
  // ---- output head: d_out = [ z1 | z1 @ Wo^T + bo ] ----
  for (int i = blockIdx.x * 256 + tid; i < Bsz * Hsz + Bsz * Osz; i += NWG * 256) {
    if (i < Bsz * Hsz) {
      out[i] = z[i];
    } else {
      int jj = i - Bsz * Hsz;
      int b = jj >> 4, o = jj & 15;
      float acc = bo[o];
      const float* zr = z + (size_t)b * Hsz;
      const float* wr = Wo + (size_t)o * Hsz;
      for (int k = 0; k < Hsz; ++k) acc += zr[k] * wr[k];
      out[i] = acc;
    }
  }
}

extern "C" void kernel_launch(void* const* d_in, const int* in_sizes, int n_in,
                              void* d_out, int out_size, void* d_ws, size_t ws_size,
                              hipStream_t stream) {
  (void)in_sizes; (void)n_in; (void)out_size; (void)ws_size;
  const float* ts = (const float*)d_in[0];
  const float* xs = (const float*)d_in[1];
  const float* z0 = (const float*)d_in[2];
  const float* W1 = (const float*)d_in[3];
  const float* b1 = (const float*)d_in[4];
  const float* W2 = (const float*)d_in[5];
  const float* b2 = (const float*)d_in[6];
  const float* W3 = (const float*)d_in[7];
  const float* b3 = (const float*)d_in[8];
  const float* Wo = (const float*)d_in[9];
  const float* bo = (const float*)d_in[10];

  char* ws = (char*)d_ws;
  h16* W1h = (h16*)ws;  ws += (size_t)Hsz * Hsz * 2;            // 32 KB
  h16* W2h = (h16*)ws;  ws += (size_t)Hsz * Hsz * 2;            // 32 KB
  h16* W3p = (h16*)ws;  ws += (size_t)Hsz * DpC * Wsz * 2;      // 1.03 MB
  float* b3p = (float*)ws; ws += (size_t)Hsz * DpC * 4;         // 16.5 KB
  float* z   = (float*)ws; ws += (size_t)Bsz * Hsz * 4;         // 64 KB
  float* kbase = (float*)ws; ws += (size_t)4 * Bsz * Hsz * 4;   // 256 KB
  unsigned* sync_ctr = (unsigned*)ws; ws += 64;

  prep_kernel<<<2112, 256, 0, stream>>>(W1, W2, W3, b3, z0,
                                        W1h, W2h, W3p, b3p, z, sync_ctr);
  cde_kernel<<<NWG, 256, 0, stream>>>(ts, xs, W1h, b1, W2h, b2, W3p, b3p,
                                      Wo, bo, z, kbase, sync_ctr, (float*)d_out);
}
